// Seq2Seq_62448824484589
// MI455X (gfx1250) — matrix operand order repaired and checked by reading, not compiled, not run on hardware
//
#include <hip/hip_runtime.h>
#include <hip/hip_bf16.h>
#include <stdint.h>
#include <stddef.h>

#define T_ENC    64
#define E_INDIM  1024
#define HDIM     1024
#define D_INDIM  512
#define VOCAB    32000
#define SENT_LEN 32

typedef __bf16 bf16;
typedef __attribute__((ext_vector_type(2)))  float v2f;
typedef __attribute__((ext_vector_type(8)))  float v8f;
typedef __attribute__((ext_vector_type(4)))  bf16  v4bf;
typedef __attribute__((ext_vector_type(8)))  bf16  v8bf;
typedef __attribute__((ext_vector_type(16))) bf16  v16bf;

__device__ __forceinline__ float sigm(float z) {
  return 1.0f / (1.0f + __expf(-z));
}

// ===========================================================================
// FP32 WMMA GEMV tile (fallback path): V_WMMA_F32_16X16X4_F32
// A (16x4): lane m holds row m K={0,1}; lane m+16 row m K={2,3}.
// B (4x16): vector broadcast across N.  Result cols identical; lane 0 has
// rows 0..7 in acc[0..7], lane 16 rows 8..15.
// ===========================================================================
__device__ __forceinline__ v8f wmma_gemv_f32(const float* __restrict__ W, int ldw,
                                             int r0, const float* __restrict__ sv,
                                             int K, int lane, v8f acc) {
  const int row   = lane & 15;
  const int khalf = (lane >> 4) << 1;
  const float* wp = W + (size_t)(r0 + row) * (size_t)ldw + khalf;
  const float* vp = sv + khalf;
#pragma unroll 4
  for (int k0 = 0; k0 < K; k0 += 4) {
    v2f a = *(const v2f*)(wp + k0);
    v2f b = *(const v2f*)(vp + k0);
    acc = __builtin_amdgcn_wmma_f32_16x16x4_f32(false, a, false, b,
                                                (short)0, acc, false, false);
  }
  return acc;
}

// ===========================================================================
// BF16 WMMA GEMV tile (fast path): V_WMMA_F32_16X16X32_BF16, fp32 accumulate.
// A (16x32 bf16): lane m (0-15) holds row m K=0..15 (v0-7);
// lane m+16 holds row m K=16..31 -> two contiguous 16B loads.
// B (32x16 bf16): vector broadcast: lanes 0-15 hold K=0..15, lanes 16-31
// K=16..31 -> one contiguous 32B LDS read per lane.
// 1 KB of weights consumed per WMMA, every byte read exactly once per wave.
// ===========================================================================
__device__ __forceinline__ v8f wmma_gemv_bf16(const bf16* __restrict__ W, int ldw,
                                              int r0, const bf16* __restrict__ sv,
                                              int K, int lane, v8f acc) {
  const int row  = lane & 15;
  const int koff = (lane >> 4) << 4;             // 0 or 16
  const bf16* wp = W + (size_t)(r0 + row) * (size_t)ldw + koff;
  const bf16* vp = sv + ((lane >> 4) << 4);      // 0 or 16
#pragma unroll 4
  for (int k0 = 0; k0 < K; k0 += 32) {
    v8bf lo = *(const v8bf*)(wp + k0);           // K chunk {0..7}+koff
    v8bf hi = *(const v8bf*)(wp + k0 + 8);       // K chunk {8..15}+koff
    v16bf a = __builtin_shufflevector(lo, hi, 0, 1, 2, 3, 4, 5, 6, 7,
                                      8, 9, 10, 11, 12, 13, 14, 15);
    v16bf b = *(const v16bf*)(vp + k0);          // 16 contiguous bf16 from LDS
    acc = __builtin_amdgcn_wmma_f32_16x16x32_bf16(false, a, false, b,
                                                  (short)0, acc, false, false);
  }
  return acc;
}

// ===========================================================================
// One LSTM time step, fp32-weight version (fallback).
// ===========================================================================
__global__ void __launch_bounds__(256)
lstm_step_f32_kernel(const float* __restrict__ xvec, const int* __restrict__ tokp,
                     const float* __restrict__ emb,
                     const float* __restrict__ Wih, const float* __restrict__ Whh,
                     const float* __restrict__ bih, const float* __restrict__ bhh,
                     const float* __restrict__ h_in, float* __restrict__ h_out,
                     float* __restrict__ c, int Kx) {
  __shared__ float sx[E_INDIM];
  __shared__ float sh[HDIM];
  __shared__ float part[8][16];

  const int tid = threadIdx.x;
  const float* xv = tokp ? (emb + (size_t)(*tokp) * (size_t)Kx) : xvec;
  for (int i = tid; i < Kx;   i += 256) sx[i] = xv[i];
  for (int i = tid; i < HDIM; i += 256) sh[i] = h_in[i];
  __syncthreads();

  const int wid = tid >> 5, lane = tid & 31;
  const int gate = wid & 3, seg = wid >> 2;
  const int r0 = gate * HDIM + blockIdx.x * 16;

  v8f acc = {};
  if (seg == 0) acc = wmma_gemv_f32(Wih, Kx,   r0, sx, Kx,   lane, acc);
  else          acc = wmma_gemv_f32(Whh, HDIM, r0, sh, HDIM, lane, acc);

  if (lane == 0) {
#pragma unroll
    for (int r = 0; r < 8; ++r) part[wid][r] = acc[r];
  } else if (lane == 16) {
#pragma unroll
    for (int r = 0; r < 8; ++r) part[wid][8 + r] = acc[r];
  }
  __syncthreads();

  if (tid < 16) {
    const int j = blockIdx.x * 16 + tid;
    const float zi = part[0][tid] + part[4][tid] + bih[0 * HDIM + j] + bhh[0 * HDIM + j];
    const float zf = part[1][tid] + part[5][tid] + bih[1 * HDIM + j] + bhh[1 * HDIM + j];
    const float zg = part[2][tid] + part[6][tid] + bih[2 * HDIM + j] + bhh[2 * HDIM + j];
    const float zo = part[3][tid] + part[7][tid] + bih[3 * HDIM + j] + bhh[3 * HDIM + j];
    const float cn = sigm(zf) * c[j] + sigm(zi) * tanhf(zg);
    c[j]     = cn;
    h_out[j] = sigm(zo) * tanhf(cn);
  }
}

// ===========================================================================
// One LSTM time step, bf16-weight version (fast path).
// ===========================================================================
__global__ void __launch_bounds__(256)
lstm_step_bf16_kernel(const float* __restrict__ xvec, const int* __restrict__ tokp,
                      const float* __restrict__ emb,
                      const bf16* __restrict__ Wih, const bf16* __restrict__ Whh,
                      const float* __restrict__ bih, const float* __restrict__ bhh,
                      const float* __restrict__ h_in, float* __restrict__ h_out,
                      float* __restrict__ c, int Kx) {
  __shared__ __align__(32) bf16 sx[E_INDIM];
  __shared__ __align__(32) bf16 sh[HDIM];
  __shared__ float part[8][16];

  const int tid = threadIdx.x;
  const float* xv = tokp ? (emb + (size_t)(*tokp) * (size_t)Kx) : xvec;
  for (int i = tid; i < Kx;   i += 256) sx[i] = (bf16)xv[i];
  for (int i = tid; i < HDIM; i += 256) sh[i] = (bf16)h_in[i];
  __syncthreads();

  const int wid = tid >> 5, lane = tid & 31;
  const int gate = wid & 3, seg = wid >> 2;
  const int r0 = gate * HDIM + blockIdx.x * 16;

  v8f acc = {};
  if (seg == 0) acc = wmma_gemv_bf16(Wih, Kx,   r0, sx, Kx,   lane, acc);
  else          acc = wmma_gemv_bf16(Whh, HDIM, r0, sh, HDIM, lane, acc);

  if (lane == 0) {
#pragma unroll
    for (int r = 0; r < 8; ++r) part[wid][r] = acc[r];
  } else if (lane == 16) {
#pragma unroll
    for (int r = 0; r < 8; ++r) part[wid][8 + r] = acc[r];
  }
  __syncthreads();

  if (tid < 16) {
    const int j = blockIdx.x * 16 + tid;
    const float zi = part[0][tid] + part[4][tid] + bih[0 * HDIM + j] + bhh[0 * HDIM + j];
    const float zf = part[1][tid] + part[5][tid] + bih[1 * HDIM + j] + bhh[1 * HDIM + j];
    const float zg = part[2][tid] + part[6][tid] + bih[2 * HDIM + j] + bhh[2 * HDIM + j];
    const float zo = part[3][tid] + part[7][tid] + bih[3 * HDIM + j] + bhh[3 * HDIM + j];
    const float cn = sigm(zf) * c[j] + sigm(zi) * tanhf(zg);
    c[j]     = cn;
    h_out[j] = sigm(zo) * tanhf(cn);
  }
}

// Pack (value, index) so atomicMax picks max value, ties -> smallest index.
__device__ __forceinline__ unsigned long long pack_key(float v, int idx) {
  unsigned u = __float_as_uint(v);
  u = (u & 0x80000000u) ? ~u : (u | 0x80000000u);
  return ((unsigned long long)u << 32) | (unsigned)(0x7FFFFFFFu - idx);
}

// ===========================================================================
// logits + greedy argmax, fp32 weights (fallback).
// ===========================================================================
__global__ void __launch_bounds__(256)
logits_argmax_f32_kernel(const float* __restrict__ h, const float* __restrict__ W,
                         const float* __restrict__ b, float* __restrict__ out,
                         unsigned long long* __restrict__ amax) {
  __shared__ float sh[HDIM];
  const int tid = threadIdx.x;
  for (int i = tid; i < HDIM; i += 256) sh[i] = h[i];
  __syncthreads();

  const int wid = tid >> 5, lane = tid & 31;
  const int j0 = blockIdx.x * 128 + wid * 16;
  v8f acc = {};
  acc = wmma_gemv_f32(W, HDIM, j0, sh, HDIM, lane, acc);

  if (lane == 0 || lane == 16) {
    const int base = j0 + ((lane >> 4) << 3);
    unsigned long long best = 0ull;
#pragma unroll
    for (int r = 0; r < 8; ++r) {
      const float v = acc[r] + b[base + r];
      out[base + r] = v;
      const unsigned long long key = pack_key(v, base + r);
      best = (key > best) ? key : best;
    }
    atomicMax(amax, best);
  }
}

// ===========================================================================
// logits + greedy argmax, bf16 weights (fast path).
// ===========================================================================
__global__ void __launch_bounds__(256)
logits_argmax_bf16_kernel(const float* __restrict__ h, const bf16* __restrict__ W,
                          const float* __restrict__ b, float* __restrict__ out,
                          unsigned long long* __restrict__ amax) {
  __shared__ __align__(32) bf16 sh[HDIM];
  const int tid = threadIdx.x;
  for (int i = tid; i < HDIM; i += 256) sh[i] = (bf16)h[i];
  __syncthreads();

  const int wid = tid >> 5, lane = tid & 31;
  const int j0 = blockIdx.x * 128 + wid * 16;
  v8f acc = {};
  acc = wmma_gemv_bf16(W, HDIM, j0, sh, HDIM, lane, acc);

  if (lane == 0 || lane == 16) {
    const int base = j0 + ((lane >> 4) << 3);
    unsigned long long best = 0ull;
#pragma unroll
    for (int r = 0; r < 8; ++r) {
      const float v = acc[r] + b[base + r];
      out[base + r] = v;
      const unsigned long long key = pack_key(v, base + r);
      best = (key > best) ? key : best;
    }
    atomicMax(amax, best);
  }
}

__global__ void finalize_kernel(unsigned long long* __restrict__ amax,
                                int* __restrict__ tok) {
  const unsigned long long key = *amax;
  *tok  = (int)(0x7FFFFFFFu - (unsigned)(key & 0xFFFFFFFFull));
  *amax = 0ull;
}

__global__ void init_kernel(float* __restrict__ h0, float* __restrict__ c,
                            int* __restrict__ tok,
                            unsigned long long* __restrict__ amax,
                            const int* __restrict__ bos) {
  const int tid = threadIdx.x;
  for (int k = tid; k < HDIM; k += 256) { h0[k] = 0.0f; c[k] = 0.0f; }
  if (tid == 0) { *tok = *bos; *amax = 0ull; }
}

// f32 -> bf16 weight conversion, float4 in / 4x bf16 out per thread-iter.
__global__ void __launch_bounds__(256)
convert_f32_bf16_kernel(const float* __restrict__ src, bf16* __restrict__ dst,
                        int n4) {
  int i = blockIdx.x * blockDim.x + threadIdx.x;
  const int stride = gridDim.x * blockDim.x;
  for (; i < n4; i += stride) {
    const float4 v = ((const float4*)src)[i];
    v4bf o = {(bf16)v.x, (bf16)v.y, (bf16)v.z, (bf16)v.w};
    ((v4bf*)dst)[i] = o;
  }
}

// ---------------------------------------------------------------------------
extern "C" void kernel_launch(void* const* d_in, const int* in_sizes, int n_in,
                              void* d_out, int out_size, void* d_ws, size_t ws_size,
                              hipStream_t stream) {
  const float* x       = (const float*)d_in[0];
  const float* enc_Wih = (const float*)d_in[1];
  const float* enc_Whh = (const float*)d_in[2];
  const float* enc_bih = (const float*)d_in[3];
  const float* enc_bhh = (const float*)d_in[4];
  const float* dec_Wih = (const float*)d_in[5];
  const float* dec_Whh = (const float*)d_in[6];
  const float* dec_bih = (const float*)d_in[7];
  const float* dec_bhh = (const float*)d_in[8];
  const float* emb     = (const float*)d_in[9];
  const float* lin_W   = (const float*)d_in[10];
  const float* lin_b   = (const float*)d_in[11];
  const int*   bos     = (const int*)d_in[12];
  float*       out     = (float*)d_out;

  // ---- workspace layout ----
  char* wsb = (char*)d_ws;
  float* hbuf[2] = {(float*)(wsb + 0), (float*)(wsb + 4096)};
  float* c       = (float*)(wsb + 8192);
  unsigned long long* amax = (unsigned long long*)(wsb + 12288);
  int*   tok     = (int*)(wsb + 12296);

  // bf16 weight region (fast path)
  const size_t N_EW  = (size_t)4 * HDIM * E_INDIM;   // enc_Wih elems
  const size_t N_EH  = (size_t)4 * HDIM * HDIM;      // enc_Whh elems
  const size_t N_DW  = (size_t)4 * HDIM * D_INDIM;   // dec_Wih elems
  const size_t N_DH  = (size_t)4 * HDIM * HDIM;      // dec_Whh elems
  const size_t N_LW  = (size_t)VOCAB * HDIM;         // lin_W elems
  const size_t OFF_W = 16384;
  bf16* enc_Wih_bf = (bf16*)(wsb + OFF_W);
  bf16* enc_Whh_bf = enc_Wih_bf + N_EW;
  bf16* dec_Wih_bf = enc_Whh_bf + N_EH;
  bf16* dec_Whh_bf = dec_Wih_bf + N_DW;
  bf16* lin_W_bf   = dec_Whh_bf + N_DH;
  const size_t need = OFF_W + 2 * (N_EW + N_EH + N_DW + N_DH + N_LW);
  const bool use_bf16 = (ws_size >= need);

  if (use_bf16) {
    // Idempotent per-call conversion (replay-safe): ~190 MB read, ~95 MB write
    // once, vs halving ~7 GB of steady-state GEMV traffic.
    convert_f32_bf16_kernel<<<2048, 256, 0, stream>>>(enc_Wih, enc_Wih_bf, (int)(N_EW / 4));
    convert_f32_bf16_kernel<<<2048, 256, 0, stream>>>(enc_Whh, enc_Whh_bf, (int)(N_EH / 4));
    convert_f32_bf16_kernel<<<2048, 256, 0, stream>>>(dec_Wih, dec_Wih_bf, (int)(N_DW / 4));
    convert_f32_bf16_kernel<<<2048, 256, 0, stream>>>(dec_Whh, dec_Whh_bf, (int)(N_DH / 4));
    convert_f32_bf16_kernel<<<4096, 256, 0, stream>>>(lin_W,   lin_W_bf,   (int)(N_LW / 4));
  }

  init_kernel<<<1, 256, 0, stream>>>(hbuf[0], c, tok, amax, bos);

  int cur = 0;
  // ---- Encoder: 64 sequential LSTM steps ----
  for (int t = 0; t < T_ENC; ++t) {
    if (use_bf16)
      lstm_step_bf16_kernel<<<HDIM / 16, 256, 0, stream>>>(
          x + (size_t)t * E_INDIM, nullptr, nullptr,
          enc_Wih_bf, enc_Whh_bf, enc_bih, enc_bhh,
          hbuf[cur], hbuf[cur ^ 1], c, E_INDIM);
    else
      lstm_step_f32_kernel<<<HDIM / 16, 256, 0, stream>>>(
          x + (size_t)t * E_INDIM, nullptr, nullptr,
          enc_Wih, enc_Whh, enc_bih, enc_bhh,
          hbuf[cur], hbuf[cur ^ 1], c, E_INDIM);
    cur ^= 1;
  }

  // ---- Decoder: 32 greedy steps ----
  for (int t = 0; t < SENT_LEN; ++t) {
    if (use_bf16)
      lstm_step_bf16_kernel<<<HDIM / 16, 256, 0, stream>>>(
          nullptr, tok, emb,
          dec_Wih_bf, dec_Whh_bf, dec_bih, dec_bhh,
          hbuf[cur], hbuf[cur ^ 1], c, D_INDIM);
    else
      lstm_step_f32_kernel<<<HDIM / 16, 256, 0, stream>>>(
          nullptr, tok, emb,
          dec_Wih, dec_Whh, dec_bih, dec_bhh,
          hbuf[cur], hbuf[cur ^ 1], c, D_INDIM);
    cur ^= 1;

    if (use_bf16)
      logits_argmax_bf16_kernel<<<VOCAB / 128, 256, 0, stream>>>(
          hbuf[cur], lin_W_bf, lin_b, out + (size_t)t * VOCAB, amax);
    else
      logits_argmax_f32_kernel<<<VOCAB / 128, 256, 0, stream>>>(
          hbuf[cur], lin_W, lin_b, out + (size_t)t * VOCAB, amax);

    finalize_kernel<<<1, 1, 0, stream>>>(amax, tok);
  }
}
